// Cross_Attn_Token_to_Image_29807073034393
// MI455X (gfx1250) — matrix operand order repaired and checked by reading, not compiled, not run on hardware
//
#include <hip/hip_runtime.h>
#include <hip/hip_bf16.h>

typedef __attribute__((ext_vector_type(8)))  _Float16 v8h;
typedef __attribute__((ext_vector_type(16))) _Float16 v16h;
typedef __attribute__((ext_vector_type(8)))  float    v8f;
typedef __attribute__((ext_vector_type(4)))  unsigned int u32x4;
typedef __attribute__((ext_vector_type(8)))  int      i32x8;
typedef __attribute__((ext_vector_type(4)))  int      i32x4;

#define DEVINL __device__ __forceinline__

#if defined(__HIP_DEVICE_COMPILE__) && __has_builtin(__builtin_amdgcn_tensor_load_to_lds) && __has_builtin(__builtin_amdgcn_s_wait_tensorcnt)
#define USE_TDM 1
#else
#define USE_TDM 0
#endif

DEVINL float fast_exp2(float x) { return __builtin_amdgcn_exp2f(x); }

DEVINL v8f wmma_f16(v16h a, v16h b, v8f c) {
  // D = A(16x32 f16) * B(32x16 f16) + C(16x16 f32)
  return __builtin_amdgcn_wmma_f32_16x16x32_f16(false, a, false, b, (short)0, c,
                                                false, false);
}

// A-matrix fragment (16x32 f16, row-major source).
// lane<16: row=lane, K=k0..k0+15
// lane>=16: row=lane-16, K=k0+16..k0+31
// Caller passes rowptr for this lane's row and koff = k0 + (lane>>4)*16.
DEVINL v16h frag_a(const _Float16* rowptr, int koff) {
  v8h lo = *(const v8h*)(rowptr + koff);
  v8h hi = *(const v8h*)(rowptr + koff + 8);
  v16h r;
#pragma unroll
  for (int i = 0; i < 8; ++i) { r[i] = lo[i]; r[i + 8] = hi[i]; }
  return r;
}

// B-matrix fragment (32x16 f16), column-major source (one row of the
// transposed matrix per lane). lane<16: col=lane, K=k0..k0+15;
// lane>=16: col=lane-16, K=k0+16..k0+31.
// Caller passes colptr for this lane's column and koff = k0 + (lane>>4)*16.
DEVINL v16h frag_b(const _Float16* colptr, int koff) {
  v8h lo = *(const v8h*)(colptr + koff);
  v8h hi = *(const v8h*)(colptr + koff + 8);
  v16h r;
#pragma unroll
  for (int i = 0; i < 8; ++i) { r[i] = lo[i]; r[i + 8] = hi[i]; }
  return r;
}

// Row-wise reductions over the C/D tile layout: each matrix row occupies one
// 16-lane half, so xor-shuffles with offsets 1,2,4,8 reduce within the row.
DEVINL v8f reduce16_max(v8f x) {
#pragma unroll
  for (int off = 1; off < 16; off <<= 1) {
#pragma unroll
    for (int r = 0; r < 8; ++r) x[r] = fmaxf(x[r], __shfl_xor(x[r], off, 32));
  }
  return x;
}
DEVINL v8f reduce16_sum(v8f x) {
#pragma unroll
  for (int off = 1; off < 16; off <<= 1) {
#pragma unroll
    for (int r = 0; r < 8; ++r) x[r] += __shfl_xor(x[r], off, 32);
  }
  return x;
}

#if USE_TDM
// Issue one 2D TDM tile load: tile_d0 elements (f16, contiguous) x tile_d1
// rows, row pitch stride0 elements. D# per CDNA5 ISA ch.8.
DEVINL void tdm_load_2d(unsigned lds_off, const _Float16* gptr,
                        unsigned tile_d0, unsigned tile_d1,
                        unsigned tensor_d0, unsigned tensor_d1,
                        unsigned stride0) {
  unsigned long long ga = (unsigned long long)gptr;  // byte address
  u32x4 g0;
  g0[0] = 1u;                     // count=1, user mode, no gather
  g0[1] = lds_off;                // lds_addr (bytes)
  g0[2] = (unsigned)ga;           // global_addr[31:0]  (bits 95:64)
  g0[3] = (unsigned)((ga >> 32) & 0x1FFFFFFu) | (2u << 30);  // addr[56:32]|type=2
  i32x8 g1;
  g1[0] = (int)(1u << 16);        // wgmask=0, data_size=1 (2B), no flags
  g1[1] = (int)((tensor_d0 & 0xFFFFu) << 16);                 // bar_addr=0, dim0.lo
  g1[2] = (int)(((tensor_d0 >> 16) & 0xFFFFu) | ((tensor_d1 & 0xFFFFu) << 16));
  g1[3] = (int)(((tensor_d1 >> 16) & 0xFFFFu) | (tile_d0 << 16));
  g1[4] = (int)(tile_d1 & 0xFFFFu);  // tile_dim1; tile_dim2=0
  g1[5] = (int)stride0;              // tensor_dim0_stride[31:0]
  g1[6] = 0;
  g1[7] = 0;
  i32x4 z4 = {0, 0, 0, 0};
#if __clang_major__ >= 23
  i32x8 z8 = {0, 0, 0, 0, 0, 0, 0, 0};
  __builtin_amdgcn_tensor_load_to_lds(g0, g1, z4, z4, z8, 0);
#else
  __builtin_amdgcn_tensor_load_to_lds(g0, g1, z4, z4, 0);
#endif
}
#endif

// ---------------------------------------------------------------------------
__global__ void k_cvt(const float* __restrict__ in, _Float16* __restrict__ out,
                      size_t n) {
  for (size_t i = (size_t)blockIdx.x * blockDim.x + threadIdx.x; i < n;
       i += (size_t)gridDim.x * blockDim.x)
    out[i] = (_Float16)in[i];
}

// Wt[n][k] = (f16) W[k][n]   (W is K x N row-major)
__global__ void k_transpose_w(const float* __restrict__ W,
                              _Float16* __restrict__ Wt, int K, int N) {
  size_t total = (size_t)K * N;
  for (size_t i = (size_t)blockIdx.x * blockDim.x + threadIdx.x; i < total;
       i += (size_t)gridDim.x * blockDim.x) {
    size_t n = i / K, k = i - n * K;
    Wt[i] = (_Float16)W[k * (size_t)N + n];
  }
}

// ---------------------------------------------------------------------------
// Generic WMMA GEMM: Out = ((A(MxK f16) * Bt(NxK f16)^T) + bias) * out_scale
// MODE 0: f16 row-major; MODE 1: f16 transposed per batch; MODE 2: f32.
// Block: 128 threads = 4 waves stacked on M. Wave tile: 32(M) x 64(N).
// grid = (M/128, N/64).
template <int MODE>
__global__ __launch_bounds__(128) void k_gemm(
    const _Float16* __restrict__ A, const _Float16* __restrict__ Bt,
    const float* __restrict__ bias, void* __restrict__ Out, int M, int N,
    int K, int rows_per_batch, float out_scale) {
  const int lane = threadIdx.x & 31;
  const int wave = threadIdx.x >> 5;
  const int lrow = lane & 15, lhi = lane >> 4;
  const int m0 = (blockIdx.x * 4 + wave) * 32;
  const int n0 = blockIdx.y * 64;

  const _Float16* aptr0 = A + (size_t)(m0 + lrow) * K;
  const _Float16* aptr1 = A + (size_t)(m0 + 16 + lrow) * K;
  const _Float16* bptr[4];
#pragma unroll
  for (int j = 0; j < 4; ++j) bptr[j] = Bt + (size_t)(n0 + j * 16 + lrow) * K;

  v8f acc[2][4] = {};
  for (int k0 = 0; k0 < K; k0 += 32) {
    v16h a0 = frag_a(aptr0, k0 + lhi * 16);
    v16h a1 = frag_a(aptr1, k0 + lhi * 16);
    v16h b[4];
#pragma unroll
    for (int j = 0; j < 4; ++j) b[j] = frag_b(bptr[j], k0 + lhi * 16);
#pragma unroll
    for (int j = 0; j < 4; ++j) acc[0][j] = wmma_f16(a0, b[j], acc[0][j]);
#pragma unroll
    for (int j = 0; j < 4; ++j) acc[1][j] = wmma_f16(a1, b[j], acc[1][j]);
  }

#pragma unroll
  for (int mi = 0; mi < 2; ++mi) {
#pragma unroll
    for (int j = 0; j < 4; ++j) {
      const int col = n0 + j * 16 + lrow;
      const float bv = bias[col];
#pragma unroll
      for (int r = 0; r < 8; ++r) {
        const int row = m0 + mi * 16 + r + lhi * 8;
        const float v = (acc[mi][j][r] + bv) * out_scale;
        if (MODE == 0) {
          ((_Float16*)Out)[(size_t)row * N + col] = (_Float16)v;
        } else if (MODE == 1) {
          const int bb = row / rows_per_batch;
          const int rb = row - bb * rows_per_batch;
          ((_Float16*)Out)[(size_t)bb * N * rows_per_batch +
                           (size_t)col * rows_per_batch + rb] = (_Float16)v;
        } else {
          ((float*)Out)[(size_t)row * N + col] = v;
        }
      }
    }
  }
}

// ---------------------------------------------------------------------------
// Dual-branch flash attention, K/V staged per block in LDS via TDM.
// Block = 256 threads (8 waves, 16 q-rows each -> q-block 128).
// The softmax scale (1/sqrt(128) * log2(e)) is pre-folded into qp/cp, so the
// softmax here is pure base-2: p = exp2(s - m).
// smem layout (bytes):
//   [0, 8192)      K tile buf0: 32 keys x 128 d (f16, row=key)
//   [8192, 16384)  K tile buf1
//   [16384, 24576) V tile buf0: 128 d x 32 keys (f16, row=d)
//   [24576, 32768) V tile buf1
//   [32768, 43008) P staging: 8 waves x 16 rows x 40 halves
#define KT_OFF 0u
#define VT_OFF 16384u
#define PST_OFF 32768u
#define SMEM_BYTES 43008

DEVINL void stage_tiles(const _Float16* __restrict__ kb,
                        const _Float16* __restrict__ vbase, int k0, int buf,
                        char* smem, int wave, int tid) {
  const unsigned kto = KT_OFF + (unsigned)buf * 8192u;
  const unsigned vto = VT_OFF + (unsigned)buf * 8192u;
#if USE_TDM
  if (wave == 0) {
    // K: 32 rows of 128 contiguous f16, pitch 2048 elements
    tdm_load_2d(kto, kb + (size_t)k0 * 2048, 128u, 32u, 2048u, 4096u, 2048u);
    // V: 128 rows of 32 contiguous f16, pitch 4096 elements
    tdm_load_2d(vto, vbase + k0, 32u, 128u, 4096u, 1024u, 4096u);
  }
#else
  (void)wave;
#pragma unroll
  for (int i = 0; i < 2; ++i) {
    const int byteoff = (tid + i * 256) * 16;
    {  // K tile: 256B per key row
      const int key = byteoff >> 8, inrow = byteoff & 255;
      v8h x = *(const v8h*)(kb + (size_t)(k0 + key) * 2048 + (inrow >> 1));
      *(v8h*)(smem + kto + byteoff) = x;
    }
    {  // V tile: 64B per d row
      const int d = byteoff >> 6, inrow = byteoff & 63;
      v8h x = *(const v8h*)(vbase + (size_t)d * 4096 + k0 + (inrow >> 1));
      *(v8h*)(smem + vto + byteoff) = x;
    }
  }
#endif
}

DEVINL void stage_fence(int wave) {
#if USE_TDM
  if (wave == 0) __builtin_amdgcn_s_wait_tensorcnt(0);
#endif
  __syncthreads();
}

// One softmax branch. arow = A row base (this lane's q/c row, pre-scaled),
// kb = K branch base for (b,h). Accumulates 0.5*softmax2(A K^T) V.
DEVINL void attn_pass(const _Float16* __restrict__ arow,
                      const _Float16* __restrict__ kb,
                      const _Float16* __restrict__ vbase, char* smem, int wave,
                      int lane, int tid, v8f* outAcc) {
  const int lrow = lane & 15, lhi = lane >> 4;
  const int NK = 4096;

  v16h af[4];
#pragma unroll
  for (int dc = 0; dc < 4; ++dc) af[dc] = frag_a(arow, dc * 32 + lhi * 16);

  v8f acc[8] = {};
  v8f m, l;
#pragma unroll
  for (int r = 0; r < 8; ++r) { m[r] = -1e30f; l[r] = 0.0f; }

  _Float16* pst = (_Float16*)(smem + PST_OFF + (unsigned)wave * 1280u);

  int buf = 0;
  stage_tiles(kb, vbase, 0, buf, smem, wave, tid);
  stage_fence(wave);

  for (int k0 = 0; k0 < NK; k0 += 32) {
    if (k0 + 32 < NK) stage_tiles(kb, vbase, k0 + 32, buf ^ 1, smem, wave, tid);

    const _Float16* ktile = (const _Float16*)(smem + KT_OFF + (unsigned)buf * 8192u);
    const _Float16* vtile = (const _Float16*)(smem + VT_OFF + (unsigned)buf * 8192u);

    // S = Q K^T. Rotating software pipeline: the fragments consumed by this
    // d-chunk's two WMMAs were loaded one iteration earlier, so the two loads
    // issued here overlap the matrix ops.
    const _Float16* krow0 = ktile + lrow * 128;
    const _Float16* krow1 = ktile + (16 + lrow) * 128;
    v8f s[2] = {};
    {
      v16h ka = frag_b(krow0, lhi * 16);
      v16h kbf = frag_b(krow1, lhi * 16);
#pragma unroll
      for (int dc = 0; dc < 4; ++dc) {
        v16h na, nb;
        if (dc < 3) {
          na = frag_b(krow0, (dc + 1) * 32 + lhi * 16);
          nb = frag_b(krow1, (dc + 1) * 32 + lhi * 16);
        }
        s[0] = wmma_f16(af[dc], ka, s[0]);
        s[1] = wmma_f16(af[dc], kbf, s[1]);
        ka = na;
        kbf = nb;
      }
    }

    // prefetch V group 0 fragments; their latency hides under the softmax
    v16h vf0[4];
#pragma unroll
    for (int jj = 0; jj < 4; ++jj)
      vf0[jj] = frag_b(vtile + (jj * 16 + lrow) * 32, lhi * 16);

    // online softmax (base-2 domain)
    v8f mx;
#pragma unroll
    for (int r = 0; r < 8; ++r) mx[r] = fmaxf(s[0][r], s[1][r]);
    mx = reduce16_max(mx);
    v8f mnew, p0, p1, rs;
    bool changed = false;
#pragma unroll
    for (int r = 0; r < 8; ++r) {
      mnew[r] = fmaxf(m[r], mx[r]);
      changed = changed || (mnew[r] > m[r]);
      p0[r] = fast_exp2(s[0][r] - mnew[r]);
      p1[r] = fast_exp2(s[1][r] - mnew[r]);
      rs[r] = p0[r] + p1[r];
    }
    rs = reduce16_sum(rs);
    // Rescale only when some row's running max actually moved (rare after
    // warm-up); wave-uniform branch.
    if (__any(changed)) {
      v8f alpha;
#pragma unroll
      for (int r = 0; r < 8; ++r) alpha[r] = fast_exp2(m[r] - mnew[r]);
#pragma unroll
      for (int r = 0; r < 8; ++r) l[r] *= alpha[r];
#pragma unroll
      for (int j = 0; j < 8; ++j)
#pragma unroll
        for (int r = 0; r < 8; ++r) acc[j][r] *= alpha[r];
    }
#pragma unroll
    for (int r = 0; r < 8; ++r) {
      l[r] += rs[r];
      m[r] = mnew[r];
    }

    // stage P (C/D layout) -> wave-private LDS -> reread as A fragment
#pragma unroll
    for (int r = 0; r < 8; ++r) {
      pst[(r + lhi * 8) * 40 + lrow] = (_Float16)p0[r];
      pst[(r + lhi * 8) * 40 + 16 + lrow] = (_Float16)p1[r];
    }
    __asm__ volatile("" ::: "memory");
    v16h pf = frag_a(pst + lrow * 40, lhi * 16);

    // O += P(16x32) * V(32x128)
    v16h vf1[4];
#pragma unroll
    for (int jj = 0; jj < 4; ++jj)
      vf1[jj] = frag_b(vtile + ((4 + jj) * 16 + lrow) * 32, lhi * 16);
#pragma unroll
    for (int jj = 0; jj < 4; ++jj) acc[jj] = wmma_f16(pf, vf0[jj], acc[jj]);
#pragma unroll
    for (int jj = 0; jj < 4; ++jj)
      acc[4 + jj] = wmma_f16(pf, vf1[jj], acc[4 + jj]);

    stage_fence(wave);
    buf ^= 1;
  }

  v8f rl;
#pragma unroll
  for (int r = 0; r < 8; ++r) rl[r] = 0.5f / l[r];
#pragma unroll
  for (int j = 0; j < 8; ++j)
#pragma unroll
    for (int r = 0; r < 8; ++r) outAcc[j][r] += acc[j][r] * rl[r];
}

// grid = (2048/128, HEADS, B); block = 256.
__global__ __launch_bounds__(256) void k_attn(
    const _Float16* __restrict__ qp, const _Float16* __restrict__ cp,
    const _Float16* __restrict__ kk, const _Float16* __restrict__ vt,
    _Float16* __restrict__ ao) {
  __shared__ __align__(16) char smem[SMEM_BYTES];
  const int tid = threadIdx.x;
  const int lane = tid & 31, wave = tid >> 5;
  const int lrow = lane & 15, lhi = lane >> 4;
  const int b = blockIdx.z, h = blockIdx.y;
  const int q0 = blockIdx.x * 128 + wave * 16;

  const _Float16* vbase = vt + ((size_t)b * 1024 + h * 128) * 4096;

  v8f outAcc[8] = {};
  attn_pass(qp + ((size_t)b * 2048 + q0 + lrow) * 1024 + h * 128,
            kk + (size_t)b * 4096 * 2048 + h * 128, vbase, smem, wave, lane,
            tid, outAcc);
  attn_pass(cp + ((size_t)b * 2048 + q0 + lrow) * 1024 + h * 128,
            kk + (size_t)b * 4096 * 2048 + 1024 + h * 128, vbase, smem, wave,
            lane, tid, outAcc);

#pragma unroll
  for (int j = 0; j < 8; ++j)
#pragma unroll
    for (int r = 0; r < 8; ++r) {
      const int row = q0 + r + lhi * 8;
      const int col = j * 16 + lrow;
      ao[((size_t)b * 2048 + row) * 1024 + h * 128 + col] =
          (_Float16)outAcc[j][r];
    }
}

// ---------------------------------------------------------------------------
extern "C" void kernel_launch(void* const* d_in, const int* in_sizes, int n_in,
                              void* d_out, int out_size, void* d_ws,
                              size_t ws_size, hipStream_t stream) {
  (void)in_sizes; (void)n_in; (void)out_size; (void)ws_size;
  const float* q   = (const float*)d_in[0];
  const float* k   = (const float*)d_in[1];
  const float* v   = (const float*)d_in[2];
  const float* cf  = (const float*)d_in[3];
  const float* Wq  = (const float*)d_in[4];
  const float* bq  = (const float*)d_in[5];
  const float* Wc  = (const float*)d_in[6];
  const float* bc  = (const float*)d_in[7];
  const float* Wk2 = (const float*)d_in[8];
  const float* bk2 = (const float*)d_in[9];
  const float* Wv  = (const float*)d_in[10];
  const float* bv  = (const float*)d_in[11];
  const float* Wo  = (const float*)d_in[12];
  const float* bo  = (const float*)d_in[13];

  _Float16* ws = (_Float16*)d_ws;
  size_t off = 0;
  auto alloc = [&](size_t n) { _Float16* p = ws + off; off += n; return p; };
  _Float16* q16  = alloc(4194304);
  _Float16* c16  = alloc(4194304);
  _Float16* k16  = alloc(8388608);
  _Float16* v16  = alloc(8388608);
  _Float16* Wqt  = alloc(1048576);
  _Float16* Wct  = alloc(1048576);
  _Float16* Wk2t = alloc(2097152);
  _Float16* Wvt  = alloc(1048576);
  _Float16* Wot  = alloc(1048576);
  _Float16* qp16 = alloc(4194304);
  _Float16* cp16 = alloc(4194304);
  _Float16* kk16 = alloc(16777216);
  _Float16* vt16 = alloc(8388608);
  _Float16* ao16 = alloc(4194304);

  k_cvt<<<dim3(2048), dim3(256), 0, stream>>>(q, q16, (size_t)4194304);
  k_cvt<<<dim3(2048), dim3(256), 0, stream>>>(cf, c16, (size_t)4194304);
  k_cvt<<<dim3(4096), dim3(256), 0, stream>>>(k, k16, (size_t)8388608);
  k_cvt<<<dim3(4096), dim3(256), 0, stream>>>(v, v16, (size_t)8388608);
  k_transpose_w<<<dim3(1024), dim3(256), 0, stream>>>(Wq, Wqt, 1024, 1024);
  k_transpose_w<<<dim3(1024), dim3(256), 0, stream>>>(Wc, Wct, 1024, 1024);
  k_transpose_w<<<dim3(2048), dim3(256), 0, stream>>>(Wk2, Wk2t, 1024, 2048);
  k_transpose_w<<<dim3(1024), dim3(256), 0, stream>>>(Wv, Wvt, 1024, 1024);
  k_transpose_w<<<dim3(1024), dim3(256), 0, stream>>>(Wo, Wot, 1024, 1024);

  // 1/sqrt(128) * log2(e): folds the softmax scale into qp/cp so the
  // attention kernel's softmax is pure exp2.
  const float s2 = 0.12751743075f;

  k_gemm<0><<<dim3(32, 16), dim3(128), 0, stream>>>(q16, Wqt, bq, qp16, 4096,
                                                    1024, 1024, 1, s2);
  k_gemm<0><<<dim3(32, 16), dim3(128), 0, stream>>>(c16, Wct, bc, cp16, 4096,
                                                    1024, 1024, 1, s2);
  k_gemm<0><<<dim3(64, 32), dim3(128), 0, stream>>>(k16, Wk2t, bk2, kk16, 8192,
                                                    2048, 1024, 1, 1.0f);
  k_gemm<1><<<dim3(64, 16), dim3(128), 0, stream>>>(v16, Wvt, bv, vt16, 8192,
                                                    1024, 1024, 4096, 1.0f);

  k_attn<<<dim3(16, 8, 2), dim3(256), 0, stream>>>(qp16, cp16, kk16, vt16,
                                                   ao16);

  k_gemm<2><<<dim3(32, 16), dim3(128), 0, stream>>>(ao16, Wot, bo,
                                                    (float*)d_out, 4096, 1024,
                                                    1024, 1, 1.0f);
}
